// DenseEquiEncoder_1683627180250
// MI455X (gfx1250) — matrix operand order repaired and checked by reading, not compiled, not run on hardware
//
#include <hip/hip_runtime.h>
#include <math.h>

// ---------------- problem constants ----------------
#define Bb 2
#define Nn_AT 512
#define NCG 64
#define Ff 128
#define N_RBF 20
#define N_CONV 3
#define Ee 32768
#define BN (Bb * Nn_AT)           // 1024
#define F3 (3 * Ff)               // 384
#define PI_F 3.14159265358979323846f
#define CUTOFF 5.0f

typedef __attribute__((ext_vector_type(16))) _Float16 v16h;
typedef __attribute__((ext_vector_type(8)))  float    v8f;

// ---------------- WMMA fragment layouts (ISA 7.12.2) ----------------
// A (16x32, 16-bit), k order used here (not ISA table): lane half h, element j -> K = j + 16*h,         row M = lane&15
// B (32x16, 16-bit), k order used here (not ISA table): lane half h, element j -> K = j + 16*h,         col N = lane&15
// C/D (16x16 f32):   reg r -> row M = r + 8*(lane>>4), col N = lane&15
//
// Packed fragment storage: frag (kchunk, ntile) is 32 lanes x 16 f16 contiguous:
//   packed[((kchunk*ntiles + ntile)*32 + lane)*16 + j]

__device__ __forceinline__ v16h load_a_frag_g(const float* __restrict__ A, int lda,
                                              int row0, int kc) {
  int l = threadIdx.x & 31;
  int half = l >> 4;
  int m = row0 + (l & 15);
  v16h a;
#pragma unroll
  for (int j = 0; j < 16; ++j) {
    int kk = kc + j + 16 * half;
    a[j] = (_Float16)A[m * lda + kk];
  }
  return a;
}

__device__ __forceinline__ float silu_f(float x) {
  return x / (1.0f + expf(-x));
}

// ---------------- weight pre-pack: f32 row-major (KxN) -> fp16 fragment order ----
__global__ void pack_w_k(const float* __restrict__ W, _Float16* __restrict__ out,
                         int K, int N) {
  int idx = blockIdx.x * blockDim.x + threadIdx.x;
  if (idx >= K * N) return;
  int j = idx & 15;
  int lane = (idx >> 4) & 31;
  int tile = idx >> 9;
  int ntiles = N >> 4;
  int ntile = tile % ntiles;
  int kchunk = tile / ntiles;
  int half = lane >> 4, l15 = lane & 15;
  int kk = kchunk * 32 + j + 16 * half;
  int n = ntile * 16 + l15;
  out[idx] = (_Float16)W[kk * N + n];
}

// ---------------- generic WMMA GEMM: C = act(A[MxK] @ W[KxN] + bias) ----------------
// one wave per 16x64 output strip (4 WMMA tiles share one A fragment).
// Wp is fragment-packed fp16.  K % 32 == 0, M % 16 == 0, N % 64 == 0.
__global__ void gemm_bias_act(const float* __restrict__ A, const _Float16* __restrict__ Wp,
                              const float* __restrict__ bias, float* __restrict__ C,
                              int M, int K, int Nc, int act) {
  int row0 = blockIdx.x * 16;
  int col0 = blockIdx.y * 64;
  int lane = threadIdx.x & 31;
  int ntiles = Nc >> 4;
  int tile0 = col0 >> 4;
  const v16h* __restrict__ Wv = (const v16h*)Wp;
  v8f cacc[4] = {{}, {}, {}, {}};
  for (int kc = 0; kc < K; kc += 32) {
    v16h a = load_a_frag_g(A, K, row0, kc);
    int kchunk = kc >> 5;
#pragma unroll
    for (int tt = 0; tt < 4; ++tt) {
      v16h b = Wv[(kchunk * ntiles + tile0 + tt) * 32 + lane];
      cacc[tt] = __builtin_amdgcn_wmma_f32_16x16x32_f16(false, a, false, b, (short)0,
                                                        cacc[tt], false, false);
    }
  }
  int half = lane >> 4;
#pragma unroll
  for (int tt = 0; tt < 4; ++tt) {
    int n = col0 + tt * 16 + (lane & 15);
    float bv = bias ? bias[n] : 0.0f;
#pragma unroll
    for (int r = 0; r < 8; ++r) {
      int m = row0 + r + 8 * half;
      float x = cacc[tt][r] + bv;
      if (act) x = silu_f(x);
      C[m * Nc + n] = x;
    }
  }
}

// ---------------- init: h_ws = h, v_ws = 0, dv = 0, out = [H_in | 0] ----------------
__global__ void init_k(const float* __restrict__ h, const float* __restrict__ Hin,
                       float* __restrict__ h_ws, float* __restrict__ v_ws,
                       float* __restrict__ dv, float* __restrict__ out) {
  int idx = blockIdx.x * blockDim.x + threadIdx.x;           // 0..393215
  if (idx >= 3 * BN * Ff) return;
  v_ws[idx] = 0.0f;
  dv[idx] = 0.0f;
  if (idx < BN * Ff) h_ws[idx] = h[idx];
  if (idx < Bb * NCG * Ff) out[idx] = Hin[idx];
  if (idx < Bb * NCG * Ff * 3) out[Bb * NCG * Ff + idx] = 0.0f;
}

// ---------------- CG geometry: d_iI, unit_iI ----------------
__global__ void geom_k(const float* __restrict__ xyz, const float* __restrict__ cg,
                       float* __restrict__ dii, float* __restrict__ unit) {
  int idx = blockIdx.x * blockDim.x + threadIdx.x;           // 0..65535 = bc*512+a
  if (idx >= Bb * NCG * Nn_AT) return;
  int a = idx & (Nn_AT - 1);
  int bc = idx >> 9;
  int b = bc >> 6, c = bc & (NCG - 1);
  float rx = xyz[(b * Nn_AT + a) * 3 + 0] - cg[(b * NCG + c) * 3 + 0];
  float ry = xyz[(b * Nn_AT + a) * 3 + 1] - cg[(b * NCG + c) * 3 + 1];
  float rz = xyz[(b * Nn_AT + a) * 3 + 2] - cg[(b * NCG + c) * 3 + 2];
  float d = sqrtf(rx * rx + ry * ry + rz * rz);
  dii[idx] = d;
  float inv = 1.0f / d;
  unit[idx * 3 + 0] = rx * inv;
  unit[idx * 3 + 1] = ry * inv;
  unit[idx * 3 + 2] = rz * inv;
}

// ---------------- edge message + scatter ----------------
__global__ void edge_msg_k(const float* __restrict__ xyz, const int* __restrict__ nbr,
                           const float* __restrict__ We, const float* __restrict__ be,
                           const float* __restrict__ phi, const float* __restrict__ v_ws,
                           float* __restrict__ h_ws, float* __restrict__ dv) {
  int e = blockIdx.x;
  int f = threadIdx.x;                                       // 0..127
  __shared__ float rbf_s[N_RBF];
  __shared__ float unit_s[3];
  __shared__ float env_s;
  __shared__ int pi_s, pj_s;
  if (f == 0) {
    int b = nbr[e * 3 + 0], i = nbr[e * 3 + 1], j = nbr[e * 3 + 2];
    pi_s = b * Nn_AT + i;
    pj_s = b * Nn_AT + j;
    float rx = xyz[(b * Nn_AT + i) * 3 + 0] - xyz[(b * Nn_AT + j) * 3 + 0];
    float ry = xyz[(b * Nn_AT + i) * 3 + 1] - xyz[(b * Nn_AT + j) * 3 + 1];
    float rz = xyz[(b * Nn_AT + i) * 3 + 2] - xyz[(b * Nn_AT + j) * 3 + 2];
    float d = sqrtf(rx * rx + ry * ry + rz * rz);
    float inv = 1.0f / d;
    unit_s[0] = rx * inv; unit_s[1] = ry * inv; unit_s[2] = rz * inv;
    env_s = (d < CUTOFF) ? 0.5f * (cosf(PI_F * d / CUTOFF) + 1.0f) : 0.0f;
    for (int r = 0; r < N_RBF; ++r)
      rbf_s[r] = sinf((float)(r + 1) * PI_F / CUTOFF * d) * inv;
  }
  __syncthreads();
  int pi = pi_s, pj = pj_s;
  float env = env_s;
  float w0 = be[f], w1 = be[Ff + f], w2 = be[2 * Ff + f];
#pragma unroll
  for (int r = 0; r < N_RBF; ++r) {
    float rb = rbf_s[r];
    w0 += rb * We[r * F3 + f];
    w1 += rb * We[r * F3 + Ff + f];
    w2 += rb * We[r * F3 + 2 * Ff + f];
  }
  float i0 = phi[pj * F3 + f] * (w0 * env);
  float i1 = phi[pj * F3 + Ff + f] * (w1 * env);
  float i2 = phi[pj * F3 + 2 * Ff + f] * (w2 * env);
  atomicAdd(&h_ws[pi * Ff + f], 0.5f * i1);
#pragma unroll
  for (int e3 = 0; e3 < 3; ++e3) {
    float contrib = 0.5f * (i2 * unit_s[e3] + i0 * v_ws[e3 * BN * Ff + pj * Ff + f]);
    atomicAdd(&dv[e3 * BN * Ff + pi * Ff + f], contrib);
  }
}

// ---------------- v += dv ; dv = 0 ----------------
__global__ void addv_k(float* __restrict__ v, float* __restrict__ dv, int n) {
  int i = blockIdx.x * blockDim.x + threadIdx.x;
  if (i < n) { v[i] += dv[i]; dv[i] = 0.0f; }
}

// ---------------- vnorm + concat ----------------
__global__ void vncat_k(const float* __restrict__ h_ws, const float* __restrict__ vv,
                        float* __restrict__ hcat) {
  int n = blockIdx.x, g = threadIdx.x;
  float acc = 1e-15f;
#pragma unroll
  for (int e3 = 0; e3 < 3; ++e3) {
    float x = vv[e3 * BN * Ff + n * Ff + g];
    acc += x * x;
  }
  hcat[n * (2 * Ff) + g] = h_ws[n * Ff + g];
  hcat[n * (2 * Ff) + Ff + g] = sqrtf(acc);
}

// ---------------- gated update apply ----------------
__global__ void upd_apply_k(float* __restrict__ h_ws, float* __restrict__ v_ws,
                            const float* __restrict__ uv, const float* __restrict__ vv,
                            const float* __restrict__ a) {
  int n = blockIdx.x, g = threadIdx.x;
  float a0 = a[n * F3 + g];
  float a1 = a[n * F3 + Ff + g];
  float a2 = a[n * F3 + 2 * Ff + g];
  float dot = 0.0f;
#pragma unroll
  for (int e3 = 0; e3 < 3; ++e3)
    dot += uv[e3 * BN * Ff + n * Ff + g] * vv[e3 * BN * Ff + n * Ff + g];
  h_ws[n * Ff + g] += 0.5f * (dot * a1 + a2);
#pragma unroll
  for (int e3 = 0; e3 < 3; ++e3)
    v_ws[e3 * BN * Ff + n * Ff + g] += 0.5f * a0 * uv[e3 * BN * Ff + n * Ff + g];
}

// ---------------- fused CG projection ----------------
// one block (256 thr = 8 waves) per (b,c). Wf cached in LDS fp16 *fragment order*,
// exp_dist A tile staged in LDS fragment order, WMMA 16x(K=128)x384, epilogue
// contracts over atoms with `assign` into LDS accumulators (k = f*3 + s interleave).
__global__ void con_fused_k(const float* __restrict__ Wf, const float* __restrict__ bfv,
                            const float* __restrict__ phic, const float* __restrict__ assign,
                            const float* __restrict__ dii, const float* __restrict__ unit,
                            const float* __restrict__ v_ws, float* __restrict__ out) {
  extern __shared__ v16h smem_v[];                        // 32B-aligned dynamic LDS
  char* smem = (char*)smem_v;
  _Float16* Ws = (_Float16*)smem;                         // 128*384 fp16 (frag order)
  _Float16* As = (_Float16*)(smem + Ff * F3 * 2);         // 16*128 fp16 (frag order)
  float* Hs  = (float*)(smem + Ff * F3 * 2 + 16 * Ff * 2);// 128 f32
  float* Vs  = Hs + Ff;                                   // 384 f32
  float* asg = Vs + F3;                                   // 16
  float* un  = asg + 16;                                  // 48
  const v16h* Wsv = (const v16h*)Ws;
  const v16h* Asv = (const v16h*)As;

  int bc = blockIdx.x;                 // 0..127
  int b = bc >> 6;
  int tid = threadIdx.x;               // 0..255
  int wave = tid >> 5, lane = tid & 31;
  int half = lane >> 4, l15 = lane & 15;

  // pack Wf into LDS in fragment order: kchunks=4, ntiles=24
  for (int i = tid; i < Ff * F3; i += 256) {
    int j = i & 15, ln = (i >> 4) & 31, tile = i >> 9;
    int ntile = tile % 24, kchunk = tile / 24;
    int hf = ln >> 4;
    int kk = kchunk * 32 + j + 16 * hf;
    int n = ntile * 16 + (ln & 15);
    Ws[i] = (_Float16)Wf[kk * F3 + n];
  }
  if (tid < Ff) Hs[tid] = 0.0f;
  for (int i = tid; i < F3; i += 256) Vs[i] = 0.0f;
  __syncthreads();

  for (int a0 = 0; a0 < Nn_AT; a0 += 16) {
    // stage exp_dist tile in A-fragment order: kchunks=4
    for (int i = tid; i < 16 * Ff; i += 256) {
      int j = i & 15, ln = (i >> 4) & 31, kchunk = i >> 9;
      int hf = ln >> 4, m = ln & 15;
      int fidx = kchunk * 32 + j + 16 * hf;
      float d = dii[bc * Nn_AT + a0 + m];
      float o = CUTOFF * (float)fidx / (float)(Ff - 1);
      float x = d - o;
      As[i] = (_Float16)expf(-x * x);
    }
    if (tid < 16) asg[tid] = assign[(b * Nn_AT + a0 + tid) * NCG + (bc & (NCG - 1))];
    if (tid < 48) un[tid] = unit[(bc * Nn_AT + a0 + tid / 3) * 3 + tid % 3];
    __syncthreads();

    if (a0 + 16 < Nn_AT)
      __builtin_prefetch(&dii[bc * Nn_AT + a0 + 16], 0, 1);

    // A fragments: one 32B LDS vector load each, shared by this wave's 3 N-tiles
    v16h afr[4];
#pragma unroll
    for (int kc4 = 0; kc4 < 4; ++kc4) afr[kc4] = Asv[kc4 * 32 + lane];

    for (int tl = 0; tl < 3; ++tl) {
      int nt = wave + tl * 8;                   // 24 N-tiles over 8 waves
      int col0 = nt * 16;
      v8f cacc = {};
#pragma unroll
      for (int kc4 = 0; kc4 < 4; ++kc4) {
        v16h bfr = Wsv[(kc4 * 24 + nt) * 32 + lane];
        cacc = __builtin_amdgcn_wmma_f32_16x16x32_f16(false, afr[kc4], false, bfr,
                                                      (short)0, cacc, false, false);
      }
      int n = col0 + l15;
      float bias = bfv[n];
      int f = n / 3, s = n - 3 * f;             // interleaved reshape (.., F, 3)
#pragma unroll
      for (int r = 0; r < 8; ++r) {
        int m = r + 8 * half;
        int a = a0 + m;
        float val = (cacc[r] + bias) * phic[(b * Nn_AT + a) * F3 + n];
        float wgt = asg[m];
        if (s == 1) {
          atomicAdd(&Hs[f], wgt * val);
        } else if (s == 2) {
#pragma unroll
          for (int e3 = 0; e3 < 3; ++e3)
            atomicAdd(&Vs[f * 3 + e3], wgt * val * un[m * 3 + e3]);
        } else {
#pragma unroll
          for (int e3 = 0; e3 < 3; ++e3)
            atomicAdd(&Vs[f * 3 + e3],
                      wgt * val * v_ws[e3 * BN * Ff + (b * Nn_AT + a) * Ff + f]);
        }
      }
    }
    __syncthreads();
  }
  // block exclusively owns (b,c): plain accumulate into out across the 3 conv launches
  if (tid < Ff) out[bc * Ff + tid] += Hs[tid];
  for (int i = tid; i < F3; i += 256)
    out[Bb * NCG * Ff + bc * F3 + i] += Vs[i];
}

// ---------------- host orchestration ----------------
extern "C" void kernel_launch(void* const* d_in, const int* in_sizes, int n_in,
                              void* d_out, int out_size, void* d_ws, size_t ws_size,
                              hipStream_t stream) {
  const float* h      = (const float*)d_in[0];
  const float* Hin    = (const float*)d_in[1];
  const float* xyz    = (const float*)d_in[2];
  const float* cgxyz  = (const float*)d_in[3];
  const float* assign = (const float*)d_in[4];
  const int*   nbr    = (const int*)d_in[6];
  const float* msg_W1 = (const float*)d_in[7];
  const float* msg_b1 = (const float*)d_in[8];
  const float* msg_W2 = (const float*)d_in[9];
  const float* msg_b2 = (const float*)d_in[10];
  const float* msg_We = (const float*)d_in[11];
  const float* msg_be = (const float*)d_in[12];
  const float* upd_U  = (const float*)d_in[13];
  const float* upd_V  = (const float*)d_in[14];
  const float* upd_W1 = (const float*)d_in[15];
  const float* upd_b1 = (const float*)d_in[16];
  const float* upd_W2 = (const float*)d_in[17];
  const float* upd_b2 = (const float*)d_in[18];
  const float* con_Wf = (const float*)d_in[19];
  const float* con_bf = (const float*)d_in[20];
  const float* con_W1 = (const float*)d_in[21];
  const float* con_b1 = (const float*)d_in[22];
  const float* con_W2 = (const float*)d_in[23];
  const float* con_b2 = (const float*)d_in[24];
  float* out = (float*)d_out;

  float* ws   = (float*)d_ws;
  float* h_ws = ws;                       // 1024*128
  float* v_ws = h_ws + BN * Ff;           // 3*1024*128
  float* dv   = v_ws + 3 * BN * Ff;       // 3*1024*128
  float* phi  = dv + 3 * BN * Ff;         // 1024*384
  float* abuf = phi + BN * F3;            // 1024*384
  float* uv   = abuf + BN * F3;           // 3*1024*128
  float* vv   = uv + 3 * BN * Ff;         // 3*1024*128
  float* tmp  = vv + 3 * BN * Ff;         // 1024*128
  float* hcat = tmp + BN * Ff;            // 1024*256
  float* dii  = hcat + BN * 2 * Ff;       // 2*64*512
  float* unit = dii + Bb * NCG * Nn_AT;   // 2*64*512*3
  _Float16* wpack = (_Float16*)(unit + Bb * NCG * Nn_AT * 3); // 3*245760 fp16

  // per-t offsets inside wpack (f16 elements)
  const size_t PT = 245760;
  const size_t oMW1 = 0, oMW2 = 16384, oUU = 65536, oUV = 81920,
               oUW1 = 98304, oUW2 = 131072, oCW1 = 180224, oCW2 = 196608;

  init_k<<<(3 * BN * Ff + 255) / 256, 256, 0, stream>>>(h, Hin, h_ws, v_ws, dv, out);
  geom_k<<<(Bb * NCG * Nn_AT + 255) / 256, 256, 0, stream>>>(xyz, cgxyz, dii, unit);

  for (int t = 0; t < N_CONV; ++t) {
    _Float16* wp = wpack + (size_t)t * PT;
    pack_w_k<<<(16384 + 255) / 256, 256, 0, stream>>>(msg_W1 + t * 16384, wp + oMW1, 128, 128);
    pack_w_k<<<(49152 + 255) / 256, 256, 0, stream>>>(msg_W2 + t * 49152, wp + oMW2, 128, 384);
    pack_w_k<<<(16384 + 255) / 256, 256, 0, stream>>>(upd_U  + t * 16384, wp + oUU,  128, 128);
    pack_w_k<<<(16384 + 255) / 256, 256, 0, stream>>>(upd_V  + t * 16384, wp + oUV,  128, 128);
    pack_w_k<<<(32768 + 255) / 256, 256, 0, stream>>>(upd_W1 + t * 32768, wp + oUW1, 256, 128);
    pack_w_k<<<(49152 + 255) / 256, 256, 0, stream>>>(upd_W2 + t * 49152, wp + oUW2, 128, 384);
    pack_w_k<<<(16384 + 255) / 256, 256, 0, stream>>>(con_W1 + t * 16384, wp + oCW1, 128, 128);
    pack_w_k<<<(49152 + 255) / 256, 256, 0, stream>>>(con_W2 + t * 49152, wp + oCW2, 128, 384);
  }

  size_t sh_con = (size_t)Ff * F3 * 2 + 16 * Ff * 2 + (Ff + F3 + 16 + 48) * 4;

  for (int t = 0; t < N_CONV; ++t) {
    _Float16* wp = wpack + (size_t)t * PT;
    // message MLP
    gemm_bias_act<<<dim3(BN / 16, Ff / 64), 32, 0, stream>>>(
        h_ws, wp + oMW1, msg_b1 + t * Ff, tmp, BN, Ff, Ff, 1);
    gemm_bias_act<<<dim3(BN / 16, F3 / 64), 32, 0, stream>>>(
        tmp, wp + oMW2, msg_b2 + t * F3, phi, BN, Ff, F3, 0);
    // edge scatter
    edge_msg_k<<<Ee, Ff, 0, stream>>>(xyz, nbr, msg_We + t * N_RBF * F3,
                                      msg_be + t * F3, phi, v_ws, h_ws, dv);
    addv_k<<<(3 * BN * Ff + 255) / 256, 256, 0, stream>>>(v_ws, dv, 3 * BN * Ff);
    // equivariant update
    gemm_bias_act<<<dim3(3 * BN / 16, Ff / 64), 32, 0, stream>>>(
        v_ws, wp + oUU, nullptr, uv, 3 * BN, Ff, Ff, 0);
    gemm_bias_act<<<dim3(3 * BN / 16, Ff / 64), 32, 0, stream>>>(
        v_ws, wp + oUV, nullptr, vv, 3 * BN, Ff, Ff, 0);
    vncat_k<<<BN, Ff, 0, stream>>>(h_ws, vv, hcat);
    gemm_bias_act<<<dim3(BN / 16, Ff / 64), 32, 0, stream>>>(
        hcat, wp + oUW1, upd_b1 + t * Ff, tmp, BN, 2 * Ff, Ff, 1);
    gemm_bias_act<<<dim3(BN / 16, F3 / 64), 32, 0, stream>>>(
        tmp, wp + oUW2, upd_b2 + t * F3, abuf, BN, Ff, F3, 0);
    upd_apply_k<<<BN, Ff, 0, stream>>>(h_ws, v_ws, uv, vv, abuf);
    // CG projection MLP + fused big GEMM/contraction
    gemm_bias_act<<<dim3(BN / 16, Ff / 64), 32, 0, stream>>>(
        h_ws, wp + oCW1, con_b1 + t * Ff, tmp, BN, Ff, Ff, 1);
    gemm_bias_act<<<dim3(BN / 16, F3 / 64), 32, 0, stream>>>(
        tmp, wp + oCW2, con_b2 + t * F3, phi, BN, Ff, F3, 0);
    con_fused_k<<<Bb * NCG, 256, sh_con, stream>>>(
        con_Wf + t * Ff * F3, con_bf + t * F3, phi, assign, dii, unit, v_ws, out);
  }
}
